// EnhancedKeyFrameSelector_89223650607541
// MI455X (gfx1250) — matrix operand order repaired and checked by reading, not compiled, not run on hardware
//
#include <hip/hip_runtime.h>
#include <math.h>

// ---------------------------------------------------------------------------
// EnhancedKeyFrameSelector on MI455X (gfx1250, wave32, WMMA + TDM).
// B=2, C=4, H=W=64, D=160 frames, K=10 selected.
// Conv pipeline via implicit-GEMM v_wmma_f32_16x16x32_bf16:
//   - A (im2col) staged to LDS with vectorized b128/b64 loads
//   - B (weights) staged to LDS with tensor_load_to_lds (TDM, async)
//   - fused BN+ReLU(+pool) epilogues avoid materializing the 671 MB tensors
// ---------------------------------------------------------------------------

typedef __bf16 v16bf __attribute__((ext_vector_type(16)));
typedef float  v8f   __attribute__((ext_vector_type(8)));
typedef unsigned int u32x4_t __attribute__((ext_vector_type(4)));
typedef int          i32x4_t __attribute__((ext_vector_type(4)));
typedef int          i32x8_t __attribute__((ext_vector_type(8)));

#define BATCH 2
#define CH    4
#define HH_   64
#define WW_   64
#define DD_   160
#define KSEL  10
#define NPOS  (BATCH * DD_ * HH_ * WW_)   // 1,310,720 spatial positions (NDHWC)

#if defined(__HIP_DEVICE_COMPILE__) && __has_builtin(__builtin_amdgcn_tensor_load_to_lds) && \
    __has_builtin(__builtin_amdgcn_s_wait_tensorcnt)
#define USE_TDM 1
#else
#define USE_TDM 0
#endif

// ---------------------------------------------------------------------------
// TDM: async DMA of a contiguous bf16 slab (global -> LDS), one issue per wave.
// Descriptor per CDNA5 ISA §8: group0 = {count|flags, lds_addr, global_addr,
// addr_hi|type=2}; group1 = {data_size=2B, tensor_dim0=n, tensor_dim1=1,
// tile_dim0=n, tile_dim1=1, dim0_stride=n}; groups 2/3 zero (<=2D tensor).
// ---------------------------------------------------------------------------
__device__ __forceinline__ void tdm_load_slab(const __bf16* gsrc, __bf16* lds_dst, int nElems) {
#if USE_TDM
  unsigned long long ga = (unsigned long long)(size_t)gsrc;
  unsigned lds = (unsigned)(size_t)lds_dst;     // flat LDS addr: low 32 bits = LDS byte offset
  unsigned ne  = (unsigned)nElems;
  u32x4_t g0 = { 1u,                                           // count=1 (valid descriptor)
                 lds,                                          // lds_addr
                 (unsigned)ga,                                 // global_addr[31:0]
                 (unsigned)((ga >> 32) & 0x01FFFFFFu) | (2u << 30) };  // addr[56:32] | type=2
  i32x8_t g1 = { (int)(1u << 16),                              // data_size=1 -> 2 bytes/elem
                 (int)((ne & 0xFFFFu) << 16),                  // tensor_dim0[15:0]
                 (int)(((ne >> 16) & 0xFFFFu) | (1u << 16)),   // tensor_dim0[31:16] | tensor_dim1=1
                 (int)(ne << 16),                              // tile_dim0 = n (fits 16b, n<=4096)
                 1,                                            // tile_dim1 = 1
                 (int)ne,                                      // tensor_dim0_stride[31:0]
                 0, 0 };
  i32x4_t g2 = { 0, 0, 0, 0 };
  i32x4_t g3 = { 0, 0, 0, 0 };
#if __clang_major__ >= 23
  i32x8_t g4 = { 0, 0, 0, 0, 0, 0, 0, 0 };
  __builtin_amdgcn_tensor_load_to_lds(g0, g1, g2, g3, g4, 0);
#else
  __builtin_amdgcn_tensor_load_to_lds(g0, g1, g2, g3, 0);
#endif
#else
  (void)gsrc; (void)lds_dst; (void)nElems;
#endif
}

// ---------------------------------------------------------------------------
// Stage 0: x [B,C,H,W,D] f32  ->  xp [B,D,H,W,C] bf16  (NDHWC for implicit GEMM)
// ---------------------------------------------------------------------------
__global__ __launch_bounds__(256) void pack_xp_kernel(const float* __restrict__ x,
                                                      __bf16* __restrict__ xp) {
  int i = blockIdx.x * 256 + threadIdx.x;
  if (i >= NPOS * CH) return;
  int c  = i & 3;
  int s  = i >> 2;
  int w  = s & 63;
  int h  = (s >> 6) & 63;
  int bd = s >> 12;
  int d  = bd % DD_;
  int b  = bd / DD_;
  xp[i] = (__bf16)x[(size_t)(((b * CH + c) * HH_ + h) * WW_ + w) * DD_ + d];
}

// ---------------------------------------------------------------------------
// Pack conv weight [O, I, 3,3,3] f32 -> [Kpad, O] bf16 with K = (kd,kh,kw,ci);
// rows k >= CIN*27 are materialized as zeros so the TDM copy needs no guards.
// ---------------------------------------------------------------------------
__global__ __launch_bounds__(256) void pack_weight_kernel(const float* __restrict__ w,
                                                          __bf16* __restrict__ out,
                                                          int CIN, int COUT, int KPAD) {
  int i = blockIdx.x * 256 + threadIdx.x;
  if (i >= KPAD * COUT) return;
  int k = i / COUT, o = i % COUT;
  float v = 0.f;
  if (k < CIN * 27) {
    int ci = k % CIN;
    int r  = k / CIN;
    int kw = r % 3, kh = (r / 3) % 3, kd = r / 9;
    v = w[(size_t)(o * CIN + ci) * 27 + kd * 9 + kh * 3 + kw];
  }
  out[i] = (__bf16)v;
}

// Fold conv-bias + eval-mode BN into per-channel scale/shift.
__global__ void prep_bn_kernel(const float* __restrict__ bias, const float* __restrict__ g,
                               const float* __restrict__ beta, const float* __restrict__ mean,
                               const float* __restrict__ var, float* __restrict__ scale,
                               float* __restrict__ shift) {
  int o = threadIdx.x;
  float s = g[o] * rsqrtf(var[o] + 1e-5f);
  scale[o] = s;
  shift[o] = (bias[o] - mean[o]) * s + beta[o];
}

__global__ __launch_bounds__(256) void zero_f32_kernel(float* __restrict__ p, int n) {
  int i = blockIdx.x * 256 + threadIdx.x;
  if (i < n) p[i] = 0.f;
}

// ---------------------------------------------------------------------------
// Implicit-GEMM 3x3x3 conv, padding 1, via WMMA bf16 -> f32.
//   A: 16 spatial positions x 32 K  (vectorized LDS stage; channel runs contiguous)
//   B: 32 K x COUT                  (TDM async copy, overlapped with A stage)
// 256-thread block = 8 waves; MT = 128/COUT M-tiles, NT = COUT/16 N-tiles.
// EPI 0: store relu(bn) bf16 NDHWC
// EPI 1: atomic mean over HxW  -> pooled [B*D, COUT]          (content conv2)
// EPI 2: atomic 16x16 avg-pool -> pooled [B*D, COUT, 4, 4]    (attn conv3)
// ---------------------------------------------------------------------------
template <int CIN, int COUT, int EPI>
__global__ __launch_bounds__(256) void conv3d_wmma_kernel(
    const __bf16* __restrict__ in, const __bf16* __restrict__ wpk,
    const float* __restrict__ bnscale, const float* __restrict__ bnshift,
    __bf16* __restrict__ out, float* __restrict__ pooled) {
  constexpr int NT     = COUT / 16;            // waves per M tile
  constexpr int MT     = 8 / NT;               // M tiles per block
  constexpr int KTOT   = CIN * 27;
  constexpr int KSTEPS = (KTOT + 31) / 32;
  constexpr int CL     = (CIN == 4) ? 2 : ((CIN == 32) ? 5 : 6);  // log2(CIN)
  constexpr int CVEC   = (CIN < 8) ? CIN : 8;  // contiguous channels per staged chunk
  constexpr int CHUNKS = 32 / CVEC;            // chunks per A row

  __shared__ __align__(16) __bf16 sA[MT * 16 * 32];
  __shared__ __align__(16) __bf16 sB[32 * COUT];

  const int tid    = threadIdx.x;
  const int lane   = tid & 31;
  const int wv     = tid >> 5;
  const int mSub   = wv / NT;
  const int nTile  = wv % NT;
  const int pBlock = blockIdx.x * (MT * 16);

  v8f acc = {};

  for (int step = 0; step < KSTEPS; ++step) {
#if USE_TDM
    if (wv == 0)   // one async DMA per block; overlaps the A-stage gather below
      tdm_load_slab(wpk + (size_t)step * 32 * COUT, sB, 32 * COUT);
#endif

    // ---- A stage: vectorized im2col gather (b128/b64 loads, b128/b64 LDS stores)
    for (int chk = tid; chk < MT * 16 * CHUNKS; chk += 256) {
      int t   = chk / (16 * CHUNKS);
      int rem = chk - t * (16 * CHUNKS);
      int m   = rem / CHUNKS;
      int kk0 = (rem - m * CHUNKS) * CVEC;
      int p   = pBlock + t * 16 + m;           // tile never crosses a W row (64 % 16 == 0)
      int w = p & 63, h = (p >> 6) & 63;
      int bd = p >> 12;
      int d = bd % DD_, b = bd / DD_;
      int k0  = step * 32 + kk0;
      int ci0 = k0 & (CIN - 1);
      int r   = k0 >> CL;
      int kw = r % 3, kh = (r / 3) % 3, kd = r / 9;
      int ww = w + kw - 1, hh = h + kh - 1, dd = d + kd - 1;
      bool ok = (k0 < KTOT) && ((unsigned)ww < 64u) && ((unsigned)hh < 64u) &&
                ((unsigned)dd < (unsigned)DD_);
      const __bf16* src = in + (size_t)(((b * DD_ + dd) * HH_ + hh) * WW_ + ww) * CIN + ci0;
      if (CVEC == 8) {
        uint4 v = {0u, 0u, 0u, 0u};
        if (ok) v = *(const uint4*)src;
        *(uint4*)&sA[t * 512 + m * 32 + kk0] = v;
      } else {
        uint2 v = {0u, 0u};
        if (ok) v = *(const uint2*)src;
        *(uint2*)&sA[t * 512 + m * 32 + kk0] = v;
      }
    }

#if !USE_TDM
    // Fallback cooperative B stage (zeros pre-materialized in wpk padding)
    for (int i = tid; i < 32 * COUT; i += 256)
      sB[i] = wpk[(size_t)step * 32 * COUT + i];
#endif

#if USE_TDM
    __builtin_amdgcn_s_wait_tensorcnt(0);      // drain TDM before anyone reads sB
#endif
    __syncthreads();

    // ---- fragments per ISA 7.12.2 wave32 layouts ----
    {
      const int half = lane >> 4;
      const int lm   = lane & 15;
      v16bf afr, bfr;
#pragma unroll
      for (int v = 0; v < 8; ++v) {
        int ka = (v < 4) ? (half * 8 + 2 * v) : (16 + half * 8 + 2 * (v - 4));
        afr[2 * v]     = sA[mSub * 512 + lm * 32 + ka];
        afr[2 * v + 1] = sA[mSub * 512 + lm * 32 + ka + 1];
        int kb = (v < 4) ? (half * 8 + 2 * v) : (16 + half * 8 + 2 * (v - 4));
        bfr[2 * v]     = sB[kb * COUT + nTile * 16 + lm];
        bfr[2 * v + 1] = sB[(kb + 1) * COUT + nTile * 16 + lm];
      }
      acc = __builtin_amdgcn_wmma_f32_16x16x32_bf16(false, afr, false, bfr,
                                                    (short)0, acc, false, false);
    }
    __syncthreads();
  }

  // ---- fused BN+ReLU(+pool) epilogue; C/D layout: lane=N, VGPR r -> M=r+8*(lane/16)
  const int half = lane >> 4;
  const int co   = nTile * 16 + (lane & 15);
  const float sc = bnscale[co], sh = bnshift[co];
#pragma unroll
  for (int r = 0; r < 8; ++r) {
    int   p   = pBlock + mSub * 16 + half * 8 + r;
    float val = acc[r] * sc + sh;
    val = val > 0.f ? val : 0.f;
    if (EPI == 0) {
      out[(size_t)p * COUT + co] = (__bf16)val;
    } else {
      int w = p & 63, h = (p >> 6) & 63, bd = p >> 12;
      if (EPI == 1) {
        atomicAdd(&pooled[(size_t)bd * COUT + co], val * (1.f / 4096.f));
      } else {
        atomicAdd(&pooled[(((size_t)bd * COUT + co) << 4) + ((h >> 4) << 2) + (w >> 4)],
                  val * (1.f / 256.f));
      }
    }
  }
}

// ---------------------------------------------------------------------------
// Area MLP on cf [B*D,128]: 128 -> 256 -> 128 -> 1  (one block per row)
// ---------------------------------------------------------------------------
__global__ __launch_bounds__(256) void content_mlp_kernel(
    const float* __restrict__ cf, const float* __restrict__ w1, const float* __restrict__ b1,
    const float* __restrict__ w2, const float* __restrict__ b2, const float* __restrict__ w3,
    const float* __restrict__ b3, float* __restrict__ scores) {
  int row = blockIdx.x, t = threadIdx.x;
  __shared__ float r0[128], r1[256], r2[128], red[128];
  if (t < 128) r0[t] = cf[(size_t)row * 128 + t];
  __syncthreads();
  {
    float a = b1[t];
    for (int i = 0; i < 128; ++i) a += r0[i] * w1[(size_t)t * 128 + i];
    r1[t] = a > 0.f ? a : 0.f;
  }
  __syncthreads();
  if (t < 128) {
    float a = b2[t];
    for (int i = 0; i < 256; ++i) a += r1[i] * w2[(size_t)t * 256 + i];
    r2[t] = a > 0.f ? a : 0.f;
  }
  __syncthreads();
  if (t < 128) red[t] = r2[t] * w3[t];
  __syncthreads();
  for (int s = 64; s > 0; s >>= 1) {
    if (t < s) red[t] += red[t + s];
    __syncthreads();
  }
  if (t == 0) scores[row] = red[0] + b3[0];
}

// ---------------------------------------------------------------------------
// Scorer MLP on af [B*D,2048]: 2048 -> 256 -> 1 -> sigmoid (one block per row)
// ---------------------------------------------------------------------------
__global__ __launch_bounds__(256) void attn_mlp_kernel(
    const float* __restrict__ af, const float* __restrict__ w1, const float* __restrict__ b1,
    const float* __restrict__ w2, const float* __restrict__ b2, float* __restrict__ scores) {
  int row = blockIdx.x, t = threadIdx.x;
  __shared__ float r0[2048];
  __shared__ float red[256];
  for (int i = t; i < 2048; i += 256) r0[i] = af[(size_t)row * 2048 + i];
  __syncthreads();
  float a = b1[t];
  const float* wr = w1 + (size_t)t * 2048;
  for (int i = 0; i < 2048; ++i) a += r0[i] * wr[i];
  a = a > 0.f ? a : 0.f;
  red[t] = a * w2[t];
  __syncthreads();
  for (int s = 128; s > 0; s >>= 1) {
    if (t < s) red[t] += red[t + s];
    __syncthreads();
  }
  if (t == 0) scores[row] = 1.f / (1.f + expf(-(red[0] + b2[0])));
}

// ---------------------------------------------------------------------------
// Frame selection: 1-D NMS (content), top-k (attn), uniform, fusion MLP,
// scatter-add vote + final top-k.  Tiny sequential work, one lane per batch.
// ---------------------------------------------------------------------------
__device__ void sort10(int* a) {
  for (int i = 1; i < KSEL; ++i) {
    int v = a[i], j = i - 1;
    while (j >= 0 && a[j] > v) { a[j + 1] = a[j]; --j; }
    a[j + 1] = v;
  }
}

__global__ __launch_bounds__(32) void select_kernel(
    const float* __restrict__ cs, const float* __restrict__ as_,
    const float* __restrict__ fw1, const float* __restrict__ fb1,
    const float* __restrict__ fw2, const float* __restrict__ fb2,
    const float* __restrict__ fw3, const float* __restrict__ fb3,
    int* __restrict__ fidx, float* __restrict__ out_idx) {
  if (threadIdx.x != 0) return;
  const int b = blockIdx.x;
  const float NEG = -3.0e38f;
  float s[DD_];
  int uni[KSEL], con[KSEL], att[KSEL], fin[KSEL];

  // ---- content: iterative argmax + window suppression (half=2) ----
  for (int i = 0; i < DD_; ++i) s[i] = cs[b * DD_ + i];
  for (int k = 0; k < KSEL; ++k) {
    int bi = 0; float bv = s[0];
    for (int i = 1; i < DD_; ++i) if (s[i] > bv) { bv = s[i]; bi = i; }
    con[k] = bi;
    int lo = bi - 2 < 0 ? 0 : bi - 2;
    int hi = bi + 2 > DD_ - 1 ? DD_ - 1 : bi + 2;
    for (int i = lo; i <= hi; ++i) s[i] = NEG;
  }
  sort10(con);

  // ---- attention: top-k ----
  for (int i = 0; i < DD_; ++i) s[i] = as_[b * DD_ + i];
  for (int k = 0; k < KSEL; ++k) {
    int bi = 0; float bv = s[0];
    for (int i = 1; i < DD_; ++i) if (s[i] > bv) { bv = s[i]; bi = i; }
    att[k] = bi;
    s[bi] = NEG;
  }
  sort10(att);

  // ---- uniform: floor(k * D/K) = 16k ----
  for (int k = 0; k < KSEL; ++k) uni[k] = k * (DD_ / KSEL);

  // ---- fusion MLP: 30 -> 128 -> 64 -> 10, softmax weights ----
  int all[3 * KSEL];
  for (int k = 0; k < KSEL; ++k) { all[k] = uni[k]; all[KSEL + k] = con[k]; all[2 * KSEL + k] = att[k]; }
  int mx = 0;
  for (int i = 0; i < 3 * KSEL; ++i) mx = all[i] > mx ? all[i] : mx;
  float Dn = (float)(mx + 1);
  float enc[3 * KSEL];
  for (int i = 0; i < 3 * KSEL; ++i) enc[i] = (float)all[i] / Dn;
  float h1[128];
  for (int j = 0; j < 128; ++j) {
    float a = fb1[j];
    for (int i = 0; i < 3 * KSEL; ++i) a += enc[i] * fw1[j * 3 * KSEL + i];
    h1[j] = a > 0.f ? a : 0.f;
  }
  float h2[64];
  for (int j = 0; j < 64; ++j) {
    float a = fb2[j];
    for (int i = 0; i < 128; ++i) a += h1[i] * fw2[j * 128 + i];
    h2[j] = a > 0.f ? a : 0.f;
  }
  float lg[KSEL];
  float lm = -3.0e38f;
  for (int j = 0; j < KSEL; ++j) {
    float a = fb3[j];
    for (int i = 0; i < 64; ++i) a += h2[i] * fw3[j * 64 + i];
    lg[j] = a;
    lm = a > lm ? a : lm;
  }
  float se = 0.f;
  for (int j = 0; j < KSEL; ++j) { lg[j] = expf(lg[j] - lm); se += lg[j]; }
  for (int j = 0; j < KSEL; ++j) lg[j] /= se;

  // ---- vote + final top-k ----
  float sc2[DD_];
  int cand[DD_];
  for (int i = 0; i < DD_; ++i) { sc2[i] = 0.f; cand[i] = 0; }
  for (int k = 0; k < KSEL; ++k) {
    sc2[uni[k]] += lg[k]; cand[uni[k]] = 1;
    sc2[con[k]] += lg[k]; cand[con[k]] = 1;
    sc2[att[k]] += lg[k]; cand[att[k]] = 1;
  }
  for (int i = 0; i < DD_; ++i) if (!cand[i]) sc2[i] = NEG;
  for (int k = 0; k < KSEL; ++k) {
    int bi = 0; float bv = sc2[0];
    for (int i = 1; i < DD_; ++i) if (sc2[i] > bv) { bv = sc2[i]; bi = i; }
    fin[k] = bi;
    sc2[bi] = NEG;
  }
  sort10(fin);
  for (int k = 0; k < KSEL; ++k) {
    fidx[b * KSEL + k]    = fin[k];
    out_idx[b * KSEL + k] = (float)fin[k];
  }
}

// ---------------------------------------------------------------------------
// Gather selected frames: selected[b,c,h,w,k] = x[b,c,h,w,fidx[b,k]]
// ---------------------------------------------------------------------------
__global__ __launch_bounds__(256) void gather_kernel(const float* __restrict__ x,
                                                     const int* __restrict__ fidx,
                                                     float* __restrict__ out) {
  int o = blockIdx.x * 256 + threadIdx.x;
  if (o >= BATCH * CH * HH_ * WW_ * KSEL) return;
  int kk = o % KSEL;
  int s2 = o / KSEL;                 // ((b*C+c)*H+h)*W+w
  int b  = s2 >> 14;                 // C*H*W = 16384
  out[o] = x[(size_t)s2 * DD_ + fidx[b * KSEL + kk]];
}

// ---------------------------------------------------------------------------
// Diversity features: per (b,k) 8x8 avg+max pooled [2C,8,8]=512, L2-normalized
// ---------------------------------------------------------------------------
__global__ __launch_bounds__(256) void feat_kernel(const float* __restrict__ sel,
                                                   float* __restrict__ fn) {
  int bk = blockIdx.x;
  int b = bk / KSEL, k = bk % KSEL;
  __shared__ float f[512];
  __shared__ float red[256];
  int t = threadIdx.x;
  for (int bin = t; bin < 512; bin += 256) {
    int cc = bin >> 6, rem = bin & 63;
    int i = rem >> 3, j = rem & 7;
    int c = cc & 3;
    float mean = 0.f, mxv = -3.0e38f;
    for (int hh = 0; hh < 8; ++hh)
      for (int ww = 0; ww < 8; ++ww) {
        int h = i * 8 + hh, w = j * 8 + ww;
        float v = sel[(size_t)(((b * CH + c) * HH_ + h) * WW_ + w) * KSEL + k];
        mean += v;
        mxv = v > mxv ? v : mxv;
      }
    f[bin] = (cc < 4) ? mean * (1.f / 64.f) : mxv;
  }
  __syncthreads();
  float p = 0.f;
  for (int bin = t; bin < 512; bin += 256) p += f[bin] * f[bin];
  red[t] = p;
  __syncthreads();
  for (int s = 128; s > 0; s >>= 1) {
    if (t < s) red[t] += red[t + s];
    __syncthreads();
  }
  float inv = 1.f / fmaxf(sqrtf(red[0]), 1e-12f);
  for (int bin = t; bin < 512; bin += 256) fn[(size_t)bk * 512 + bin] = f[bin] * inv;
}

__global__ __launch_bounds__(256) void loss_kernel(const float* __restrict__ fn,
                                                   float* __restrict__ out) {
  __shared__ float red[256];
  int t = threadIdx.x;
  float p = 0.f;
  for (int pr = t; pr < BATCH * KSEL * KSEL; pr += 256) {
    int b = pr / (KSEL * KSEL);
    int rem = pr % (KSEL * KSEL);
    int k = rem / KSEL, l = rem % KSEL;
    if (k == l) continue;
    const float* A = fn + (size_t)(b * KSEL + k) * 512;
    const float* Bv = fn + (size_t)(b * KSEL + l) * 512;
    float d = 0.f;
    for (int i = 0; i < 512; ++i) d += A[i] * Bv[i];
    p += d;
  }
  red[t] = p;
  __syncthreads();
  for (int s = 128; s > 0; s >>= 1) {
    if (t < s) red[t] += red[t + s];
    __syncthreads();
  }
  if (t == 0) out[0] = red[0] / (float)(BATCH * KSEL * (KSEL - 1));
}

// ---------------------------------------------------------------------------
// Host launcher
// ---------------------------------------------------------------------------
extern "C" void kernel_launch(void* const* d_in, const int* in_sizes, int n_in,
                              void* d_out, int out_size, void* d_ws, size_t ws_size,
                              hipStream_t stream) {
  (void)in_sizes; (void)n_in; (void)out_size; (void)ws_size;
  const float* x = (const float*)d_in[0];
  // content
  const float* c_w1 = (const float*)d_in[1];
  const float* c_b1 = (const float*)d_in[2];
  const float* c_bn1g = (const float*)d_in[3], *c_bn1b = (const float*)d_in[4];
  const float* c_bn1m = (const float*)d_in[5], *c_bn1v = (const float*)d_in[6];
  const float* c_w2 = (const float*)d_in[7];
  const float* c_b2 = (const float*)d_in[8];
  const float* c_bn2g = (const float*)d_in[9], *c_bn2b = (const float*)d_in[10];
  const float* c_bn2m = (const float*)d_in[11], *c_bn2v = (const float*)d_in[12];
  // area
  const float* ar_w1 = (const float*)d_in[13], *ar_b1 = (const float*)d_in[14];
  const float* ar_w2 = (const float*)d_in[15], *ar_b2 = (const float*)d_in[16];
  const float* ar_w3 = (const float*)d_in[17], *ar_b3 = (const float*)d_in[18];
  // attn
  const float* a_w1 = (const float*)d_in[19];
  const float* a_b1 = (const float*)d_in[20];
  const float* a_bn1g = (const float*)d_in[21], *a_bn1b = (const float*)d_in[22];
  const float* a_bn1m = (const float*)d_in[23], *a_bn1v = (const float*)d_in[24];
  const float* a_w2 = (const float*)d_in[25];
  const float* a_b2 = (const float*)d_in[26];
  const float* a_bn2g = (const float*)d_in[27], *a_bn2b = (const float*)d_in[28];
  const float* a_bn2m = (const float*)d_in[29], *a_bn2v = (const float*)d_in[30];
  const float* a_w3 = (const float*)d_in[31];
  const float* a_b3 = (const float*)d_in[32];
  const float* a_bn3g = (const float*)d_in[33], *a_bn3b = (const float*)d_in[34];
  const float* a_bn3m = (const float*)d_in[35], *a_bn3v = (const float*)d_in[36];
  // scorer
  const float* s_w1 = (const float*)d_in[37], *s_b1 = (const float*)d_in[38];
  const float* s_w2 = (const float*)d_in[39], *s_b2 = (const float*)d_in[40];
  // fusion
  const float* f_w1 = (const float*)d_in[41], *f_b1 = (const float*)d_in[42];
  const float* f_w2 = (const float*)d_in[43], *f_b2 = (const float*)d_in[44];
  const float* f_w3 = (const float*)d_in[45], *f_b3 = (const float*)d_in[46];

  float* out = (float*)d_out;  // [selected 327680][final_idx 20][div_loss 1]

  // ---- workspace carve (peak ~268 MB) ----
  char* ws = (char*)d_ws;
  size_t off = 0;
  auto carve = [&](size_t bytes) -> char* {
    char* p = ws + off;
    off += (bytes + 255) & ~(size_t)255;
    return p;
  };
  __bf16* xp    = (__bf16*)carve((size_t)NPOS * CH * 2);   // 10.5 MB
  __bf16* act64 = (__bf16*)carve((size_t)NPOS * 64 * 2);   // 167.8 MB (content h1, reused attn g2)
  __bf16* act32 = (__bf16*)carve((size_t)NPOS * 32 * 2);   // 83.9 MB  (attn g1)
  __bf16* wc1 = (__bf16*)carve(128 * 64 * 2);
  __bf16* wc2 = (__bf16*)carve(1728 * 128 * 2);
  __bf16* wa1 = (__bf16*)carve(128 * 32 * 2);
  __bf16* wa2 = (__bf16*)carve(864 * 64 * 2);
  __bf16* wa3 = (__bf16*)carve(1728 * 128 * 2);
  float* scC1 = (float*)carve(64 * 4);  float* shC1 = (float*)carve(64 * 4);
  float* scC2 = (float*)carve(128 * 4); float* shC2 = (float*)carve(128 * 4);
  float* scA1 = (float*)carve(32 * 4);  float* shA1 = (float*)carve(32 * 4);
  float* scA2 = (float*)carve(64 * 4);  float* shA2 = (float*)carve(64 * 4);
  float* scA3 = (float*)carve(128 * 4); float* shA3 = (float*)carve(128 * 4);
  float* cf  = (float*)carve((size_t)BATCH * DD_ * 128 * 4);
  float* af  = (float*)carve((size_t)BATCH * DD_ * 2048 * 4);
  float* csc = (float*)carve((size_t)BATCH * DD_ * 4);
  float* asc = (float*)carve((size_t)BATCH * DD_ * 4);
  int*   fidx = (int*)carve(BATCH * KSEL * 4);
  float* fn  = (float*)carve((size_t)BATCH * KSEL * 512 * 4);

  // ---- prep ----
  pack_xp_kernel<<<(NPOS * CH) / 256, 256, 0, stream>>>(x, xp);
  pack_weight_kernel<<<(128 * 64 + 255) / 256, 256, 0, stream>>>(c_w1, wc1, 4, 64, 128);
  pack_weight_kernel<<<(1728 * 128 + 255) / 256, 256, 0, stream>>>(c_w2, wc2, 64, 128, 1728);
  pack_weight_kernel<<<(128 * 32 + 255) / 256, 256, 0, stream>>>(a_w1, wa1, 4, 32, 128);
  pack_weight_kernel<<<(864 * 64 + 255) / 256, 256, 0, stream>>>(a_w2, wa2, 32, 64, 864);
  pack_weight_kernel<<<(1728 * 128 + 255) / 256, 256, 0, stream>>>(a_w3, wa3, 64, 128, 1728);
  prep_bn_kernel<<<1, 64, 0, stream>>>(c_b1, c_bn1g, c_bn1b, c_bn1m, c_bn1v, scC1, shC1);
  prep_bn_kernel<<<1, 128, 0, stream>>>(c_b2, c_bn2g, c_bn2b, c_bn2m, c_bn2v, scC2, shC2);
  prep_bn_kernel<<<1, 32, 0, stream>>>(a_b1, a_bn1g, a_bn1b, a_bn1m, a_bn1v, scA1, shA1);
  prep_bn_kernel<<<1, 64, 0, stream>>>(a_b2, a_bn2g, a_bn2b, a_bn2m, a_bn2v, scA2, shA2);
  prep_bn_kernel<<<1, 128, 0, stream>>>(a_b3, a_bn3g, a_bn3b, a_bn3m, a_bn3v, scA3, shA3);
  zero_f32_kernel<<<(BATCH * DD_ * 128 + 255) / 256, 256, 0, stream>>>(cf, BATCH * DD_ * 128);
  zero_f32_kernel<<<(BATCH * DD_ * 2048 + 255) / 256, 256, 0, stream>>>(af, BATCH * DD_ * 2048);

  // ---- content analyzer: conv(4->64) -> conv(64->128)+mean(H,W) ----
  conv3d_wmma_kernel<4, 64, 0><<<NPOS / 32, 256, 0, stream>>>(xp, wc1, scC1, shC1, act64, (float*)nullptr);
  conv3d_wmma_kernel<64, 128, 1><<<NPOS / 16, 256, 0, stream>>>(act64, wc2, scC2, shC2, (__bf16*)nullptr, cf);
  content_mlp_kernel<<<BATCH * DD_, 256, 0, stream>>>(cf, ar_w1, ar_b1, ar_w2, ar_b2, ar_w3, ar_b3, csc);

  // ---- attention extractor: conv(4->32) -> conv(32->64) -> conv(64->128)+pool4x4 ----
  conv3d_wmma_kernel<4, 32, 0><<<NPOS / 64, 256, 0, stream>>>(xp, wa1, scA1, shA1, act32, (float*)nullptr);
  conv3d_wmma_kernel<32, 64, 0><<<NPOS / 32, 256, 0, stream>>>(act32, wa2, scA2, shA2, act64, (float*)nullptr);
  conv3d_wmma_kernel<64, 128, 2><<<NPOS / 16, 256, 0, stream>>>(act64, wa3, scA3, shA3, (__bf16*)nullptr, af);
  attn_mlp_kernel<<<BATCH * DD_, 256, 0, stream>>>(af, s_w1, s_b1, s_w2, s_b2, asc);

  // ---- selection, gather, diversity ----
  select_kernel<<<BATCH, 32, 0, stream>>>(csc, asc, f_w1, f_b1, f_w2, f_b2, f_w3, f_b3,
                                          fidx, out + BATCH * CH * HH_ * WW_ * KSEL);
  gather_kernel<<<(BATCH * CH * HH_ * WW_ * KSEL + 255) / 256, 256, 0, stream>>>(x, fidx, out);
  feat_kernel<<<BATCH * KSEL, 256, 0, stream>>>(out, fn);
  loss_kernel<<<1, 256, 0, stream>>>(fn, out + BATCH * CH * HH_ * WW_ * KSEL + BATCH * KSEL);
}
